// StructureDecoder_28140625723763
// MI455X (gfx1250) — matrix operand order repaired and checked by reading, not compiled, not run on hardware
//
#include <hip/hip_runtime.h>

// ---------------------------------------------------------------------------
// StructureDecoder for MI455X (gfx1250, wave32, WMMA)
//
// Pipeline: proj GEMM -> edge scatter-add (f32 global atomics) ->
//           bias/ReLU/threefry-dropout/f16-convert -> x @ x^T via
//           v_wmma_f32_16x16x32_f16 (f32 accumulate).
// The gram matrix writes 1.07 GB; x (2 MB f16) lives in L2, so with f16 WMMA
// the kernel is output-store bound (~46 us at 23.3 TB/s).
// ---------------------------------------------------------------------------

typedef __attribute__((ext_vector_type(16))) _Float16 v16h;
typedef __attribute__((ext_vector_type(8)))  _Float16 v8h;
typedef __attribute__((ext_vector_type(8)))  float    v8f;

#define N_NODES 16384
#define NUM_E   524288
#define IN_FEAT 128
#define OUT_FEAT 64
#define KEEP_P  0.7f

// ------------------------------- 1) hw = h @ W ------------------------------
__global__ __launch_bounds__(256) void k_proj(const float* __restrict__ h,
                                              const float* __restrict__ w,
                                              float* __restrict__ hw) {
    int tid = threadIdx.x;
    int o   = tid & 63;
    int row = blockIdx.x * 4 + (tid >> 6);
    const float* hr = h + (size_t)row * IN_FEAT;
    float acc = 0.f;
#pragma unroll 8
    for (int k = 0; k < IN_FEAT; ++k)
        acc = __fmaf_rn(hr[k], w[k * OUT_FEAT + o], acc);
    hw[(size_t)row * OUT_FEAT + o] = acc;
}

// ------------------------------- 2) zero agg --------------------------------
__global__ __launch_bounds__(256) void k_zero(float* __restrict__ p) {
    p[blockIdx.x * 256 + threadIdx.x] = 0.f;
}

// ------------------- 3) agg[dst] += hw[src] (scatter-add) -------------------
__global__ __launch_bounds__(256) void k_scatter(const float* __restrict__ hw,
                                                 const int* __restrict__ src,
                                                 const int* __restrict__ dst,
                                                 float* __restrict__ agg) {
    int t = blockIdx.x * 256 + threadIdx.x;   // t < NUM_E * 64
    int e = t >> 6;
    int f = t & 63;
    int s = src[e];
    int d = dst[e];
    float v = hw[(size_t)s * OUT_FEAT + f];
    unsafeAtomicAdd(&agg[(size_t)d * OUT_FEAT + f], v);  // global_atomic_add_f32
}

// -------- 4) x = dropout(relu(agg + bias)) -> f16, JAX threefry2x32 ---------
__device__ __forceinline__ unsigned tf_rotl(unsigned x, unsigned r) {
    return (x << r) | (x >> (32u - r));
}

__device__ __forceinline__ void threefry2x32(unsigned k0, unsigned k1,
                                             unsigned& x0, unsigned& x1) {
    unsigned k2 = k0 ^ k1 ^ 0x1BD11BDAu;
    x0 += k0; x1 += k1;
#define TF_R(r) { x0 += x1; x1 = tf_rotl(x1, r); x1 ^= x0; }
    TF_R(13) TF_R(15) TF_R(26) TF_R(6)
    x0 += k1; x1 += k2 + 1u;
    TF_R(17) TF_R(29) TF_R(16) TF_R(24)
    x0 += k2; x1 += k0 + 2u;
    TF_R(13) TF_R(15) TF_R(26) TF_R(6)
    x0 += k0; x1 += k1 + 3u;
    TF_R(17) TF_R(29) TF_R(16) TF_R(24)
    x0 += k1; x1 += k2 + 4u;
    TF_R(13) TF_R(15) TF_R(26) TF_R(6)
    x0 += k2; x1 += k0 + 5u;
#undef TF_R
}

__global__ __launch_bounds__(256) void k_act(const float* __restrict__ agg,
                                             const float* __restrict__ bias,
                                             _Float16* __restrict__ xh) {
    unsigned idx = blockIdx.x * 256 + threadIdx.x;     // < N_NODES*OUT_FEAT
    float v = agg[idx] + bias[idx & 63];
    v = v > 0.f ? v : 0.f;

    // jax.random.bernoulli(jax.random.key(42), 0.7, (N,64)):
    // key = (0, 42); counts = iota(n) split into halves (x0=i, x1=i+n/2).
    const unsigned half = (N_NODES * OUT_FEAT) / 2u;
    unsigned pi = idx < half ? idx : idx - half;
    unsigned x0 = pi, x1 = pi + half;
    threefry2x32(0u, 42u, x0, x1);
    unsigned bits = idx < half ? x0 : x1;
    float u = __uint_as_float((bits >> 9) | 0x3f800000u) - 1.0f;  // U[0,1)
    v = (u < KEEP_P) ? v * (1.0f / KEEP_P) : 0.0f;

    xh[idx] = (_Float16)v;
}

// ------------------------ 5) out = x @ x^T via WMMA -------------------------
// A fragment (16x32 f16, MxK), rows row_base..+15, K = kb..kb+31.
// ISA layout: lane<16 -> M=lane, K {kb+0..7, kb+16..23}; lane>=16 -> M=lane-16,
// K {kb+8..15, kb+24..31}. Two contiguous 16B loads per lane.
__device__ __forceinline__ v16h load_a_frag(const _Float16* __restrict__ xh,
                                            int row_base, int kb, int lane) {
    int m  = lane & 15;
    int hi = lane >> 4;
    const _Float16* p = xh + (size_t)(row_base + m) * OUT_FEAT + kb + hi * 8;
    v8h lo  = *(const v8h*)p;
    v8h hi8 = *(const v8h*)(p + 16);
    return __builtin_shufflevector(lo, hi8, 0,1,2,3,4,5,6,7,8,9,10,11,12,13,14,15);
}

// B fragment (32x16 f16, KxN) for out columns c0..c0+15: B[k][n] = x[c0+n][k].
// ISA layout: lane -> column N=lane%16; lanes 0-15 hold K {kb+0..7, kb+16..23},
// lanes 16-31 hold K {kb+8..15, kb+24..31}. Two contiguous 16B loads per lane.
__device__ __forceinline__ v16h load_b_frag(const _Float16* __restrict__ xh,
                                            int col_base, int kb, int lane) {
    int n  = lane & 15;
    int hi = lane >> 4;
    const _Float16* p = xh + (size_t)(col_base + n) * OUT_FEAT + kb + hi * 8;
    v8h lo  = *(const v8h*)p;
    v8h hi8 = *(const v8h*)(p + 16);
    return __builtin_shufflevector(lo, hi8, 0,1,2,3,4,5,6,7,8,9,10,11,12,13,14,15);
}

__global__ __launch_bounds__(256) void k_gram(const _Float16* __restrict__ xh,
                                              float* __restrict__ out) {
    int lane = threadIdx.x & 31;
    int wave = threadIdx.x >> 5;
    int row_base = blockIdx.y * 16;                 // 16 output rows per wave
    int col_base = blockIdx.x * 512 + wave * 64;    // 64 output cols per wave

    // A fragments reused across the 4 column tiles of this wave.
    v16h a0 = load_a_frag(xh, row_base, 0,  lane);
    v16h a1 = load_a_frag(xh, row_base, 32, lane);

    int n  = lane & 15;
    int hi = lane >> 4;

#pragma unroll
    for (int t = 0; t < 4; ++t) {
        int c0 = col_base + t * 16;
        v16h b0 = load_b_frag(xh, c0, 0,  lane);
        v16h b1 = load_b_frag(xh, c0, 32, lane);
        v8f acc = {};
        acc = __builtin_amdgcn_wmma_f32_16x16x32_f16(
            false, a0, false, b0, (short)0, acc, false, false);
        acc = __builtin_amdgcn_wmma_f32_16x16x32_f16(
            false, a1, false, b1, (short)0, acc, false, false);

        // D layout: lane holds column n = lane%16; VGPR r -> row r + 8*(lane/16).
        float* op = out + (size_t)(row_base + hi * 8) * N_NODES + c0 + n;
#pragma unroll
        for (int r = 0; r < 8; ++r)
            op[(size_t)r * N_NODES] = acc[r];
    }
}

// ---------------------------------------------------------------------------
extern "C" void kernel_launch(void* const* d_in, const int* in_sizes, int n_in,
                              void* d_out, int out_size, void* d_ws, size_t ws_size,
                              hipStream_t stream) {
    const float* h    = (const float*)d_in[0];
    const int*   src  = (const int*)d_in[1];
    const int*   dst  = (const int*)d_in[2];
    const float* w    = (const float*)d_in[3];
    const float* bias = (const float*)d_in[4];
    float* out = (float*)d_out;

    char* ws = (char*)d_ws;
    float*    hw  = (float*)ws;                         // 4 MB: [N,64] f32
    float*    agg = (float*)(ws + (4u << 20));          // 4 MB: [N,64] f32
    _Float16* xh  = (_Float16*)(ws + (8u << 20));       // 2 MB: [N,64] f16

    k_proj   <<<N_NODES / 4, 256, 0, stream>>>(h, w, hw);
    k_zero   <<<(N_NODES * OUT_FEAT) / 256, 256, 0, stream>>>(agg);
    k_scatter<<<(NUM_E * 64) / 256, 256, 0, stream>>>(hw, src, dst, agg);
    k_act    <<<(N_NODES * OUT_FEAT) / 256, 256, 0, stream>>>(agg, bias, xh);
    k_gram   <<<dim3(N_NODES / 512, N_NODES / 16), 256, 0, stream>>>(xh, out);
}
